// funcdecllstm_25305947308889
// MI455X (gfx1250) — matrix operand order repaired and checked by reading, not compiled, not run on hardware
//
#include <hip/hip_runtime.h>
#include <math.h>

#define MEMD 256          // hidden size
#define KP   264          // padded bf16 row stride for Wfh in LDS (avoids 16-way bank conflict)
#define ROWS 256          // children rows per block
#define NWAVE 8           // 256 threads = 8 wave32
#define MT 2              // 16-row M-tiles per wave

typedef __attribute__((ext_vector_type(16))) __bf16        v16bf;
typedef __attribute__((ext_vector_type(8)))  float          v8f;
typedef __attribute__((ext_vector_type(16))) unsigned short us16;
typedef __attribute__((ext_vector_type(8)))  unsigned short us8;
typedef __attribute__((ext_vector_type(4)))  float          f4;

__device__ __forceinline__ unsigned short f2bf(float f) {
  // fp32 -> bf16 round-to-nearest-even
  unsigned u = __builtin_bit_cast(unsigned, f);
  u += 0x7FFFu + ((u >> 16) & 1u);
  return (unsigned short)(u >> 16);
}

// Fast sigmoid: v_exp_f32 + v_rcp_f32 (approximate reciprocal, ~1 ulp) —
// avoids the IEEE-correct v_div_scale/v_div_fmas sequence in the hot epilogue.
__device__ __forceinline__ float sigmoid_fast(float x) {
  return __builtin_amdgcn_rcpf(1.0f + __expf(-x));
}

__device__ __forceinline__ float sigmoidf(float x) {
  return 1.0f / (1.0f + __expf(-x));
}

// ---------------------------------------------------------------------------
// Kernel A: gfx[m] = inputs . Wfx[m,:] + bfx[m] + bfh[m]   (one block, 256 thr)
// ---------------------------------------------------------------------------
__global__ void tree_lstm_gfx_kernel(const float* __restrict__ inputs,
                                     const float* __restrict__ Wfx,
                                     const float* __restrict__ bfx,
                                     const float* __restrict__ bfh,
                                     float* __restrict__ gfx) {
  __shared__ float xin[MEMD];
  const int m = threadIdx.x;
  xin[m] = inputs[m];
  __syncthreads();
  const float* w = Wfx + m * MEMD;
  float acc = 0.0f;
#pragma unroll 8
  for (int k = 0; k < MEMD; ++k) acc += xin[k] * w[k];
  gfx[m] = acc + bfx[m] + bfh[m];
}

// ---------------------------------------------------------------------------
// Kernel B: per-child forget gates via bf16 WMMA + weighted reduction.
//   partial[block][m] = sum_{rows in block} sigmoid((child_h @ Wfh^T)[r,m] + gfx[m]) * child_c[r,m]
// Tail handling: row indices are CLAMPED (loads always in-bounds, no exec
// divergence); the bounds mask is applied once at the f*child_c product.
// ---------------------------------------------------------------------------
__global__ void __launch_bounds__(256)
tree_lstm_forget_kernel(const float* __restrict__ child_h,
                        const float* __restrict__ child_c,
                        const float* __restrict__ Wfh,
                        const float* __restrict__ gfx,
                        float* __restrict__ partial,
                        int nch) {
  extern __shared__ char smem[];
  unsigned short* Wb    = (unsigned short*)smem;                      // [256][KP] bf16
  float*          wpart = (float*)(smem + (size_t)MEMD * KP * 2);     // [NWAVE][256]
  float*          gl    = wpart + NWAVE * MEMD;                       // [256]

  const int tid = threadIdx.x;

  // Stage Wfh fp32 -> bf16 into LDS (coalesced reads from global).
  for (int i = tid; i < MEMD * MEMD; i += 256) {
    const int m = i >> 8, k = i & 255;
    Wb[m * KP + k] = f2bf(Wfh[i]);
  }
  for (int i = tid; i < NWAVE * MEMD; i += 256) wpart[i] = 0.0f;
  gl[tid] = gfx[tid];
  __syncthreads();

  const int wave = tid >> 5;
  const int lane = tid & 31;
  const int half = lane >> 4;   // which 16-lane half of the wave
  const int l16  = lane & 15;
  const long blockrow = (long)blockIdx.x * ROWS;
  const long rmax = (long)nch - 1;

  for (int mt = 0; mt < MT; ++mt) {
    const long base = blockrow + (long)wave * (MT * 16) + (long)mt * 16;

    // ---- Build A fragments (16x32 bf16 per K-step) for the whole K=256 strip.
    // A layout: lanes 0-15 row M=lane, K = {0..7,16..23}+32s ;
    //           lanes 16-31 row M=lane-16, K = {8..15,24..31}+32s.
    const long arow = base + l16;
    const long arc  = arow < rmax ? arow : rmax;     // clamped: always valid
    const float* hrow = child_h + arc * MEMD;
    v16bf a[8];
#pragma unroll
    for (int s = 0; s < 8; ++s) {
      const int kb = s * 32 + half * 8;
      const f4 c0 = *(const f4*)(hrow + kb);
      const f4 c1 = *(const f4*)(hrow + kb + 4);
      const f4 c2 = *(const f4*)(hrow + kb + 16);
      const f4 c3 = *(const f4*)(hrow + kb + 20);
      us16 u;
      u[0]  = f2bf(c0.x); u[1]  = f2bf(c0.y); u[2]  = f2bf(c0.z); u[3]  = f2bf(c0.w);
      u[4]  = f2bf(c1.x); u[5]  = f2bf(c1.y); u[6]  = f2bf(c1.z); u[7]  = f2bf(c1.w);
      u[8]  = f2bf(c2.x); u[9]  = f2bf(c2.y); u[10] = f2bf(c2.z); u[11] = f2bf(c2.w);
      u[12] = f2bf(c3.x); u[13] = f2bf(c3.y); u[14] = f2bf(c3.z); u[15] = f2bf(c3.w);
      a[s] = __builtin_bit_cast(v16bf, u);
    }

    // ---- Sweep all 16 N-tiles reusing the register-resident A strip.
    for (int nt = 0; nt < 16; ++nt) {
      const int m = nt * 16 + l16;
      // B layout (32x16 bf16): lane column n=l16; lanes 0-15 K={0..7,16..23},
      // lanes 16-31 K={8..15,24..31} of each 32-K window; 2 packed K per VGPR.
      const unsigned short* bp = Wb + m * KP + half * 8;
      v8f acc = {};
#pragma unroll
      for (int s = 0; s < 8; ++s) {
        us8 b0 = *(const us8*)(bp + s * 32);
        us8 b1 = *(const us8*)(bp + s * 32 + 16);
        us16 ub = __builtin_shufflevector(b0, b1,
                    0, 1, 2, 3, 4, 5, 6, 7, 8, 9, 10, 11, 12, 13, 14, 15);
        v16bf b = __builtin_bit_cast(v16bf, ub);
        acc = __builtin_amdgcn_wmma_f32_16x16x32_bf16(
                  false, a[s], false, b, (short)0, acc, false, false);
      }

      // D layout: VGPR j, lane -> element (row = j + 8*half, col = l16).
      const float g = gl[m];
      float ssum = 0.0f;
#pragma unroll
      for (int j = 0; j < 8; ++j) {
        const long r  = base + j + half * 8;
        const long rc = r < rmax ? r : rmax;         // clamped: load always valid
        const float f  = sigmoid_fast(acc[j] + g);
        const float cc = child_c[rc * MEMD + m];     // unconditional, coalesced
        ssum += (r <= rmax) ? f * cc : 0.0f;         // single v_cndmask per j
      }
      // lanes L and L+16 hold the same output column m -> combine halves.
      ssum += __shfl_xor(ssum, 16, 32);
      if (half == 0) wpart[wave * MEMD + m] += ssum;
    }
  }

  __syncthreads();
  // Deterministic fixed-order reduction across the 8 waves.
  float s = 0.0f;
#pragma unroll
  for (int w = 0; w < NWAVE; ++w) s += wpart[w * MEMD + tid];
  partial[(long)blockIdx.x * MEMD + tid] = s;
}

// ---------------------------------------------------------------------------
// Kernel C: i/o/u gates, fixed-order partial reduction, c & h.
// ---------------------------------------------------------------------------
__global__ void tree_lstm_final_kernel(const float* __restrict__ inputs,
                                       const float* __restrict__ hsum,
                                       const float* __restrict__ Wix, const float* __restrict__ bix,
                                       const float* __restrict__ Wux, const float* __restrict__ bux,
                                       const float* __restrict__ Wox, const float* __restrict__ box,
                                       const float* __restrict__ Wih, const float* __restrict__ bih,
                                       const float* __restrict__ Wuh, const float* __restrict__ buh,
                                       const float* __restrict__ Woh, const float* __restrict__ boh,
                                       const float* __restrict__ partial, int nblk,
                                       float* __restrict__ out) {
  __shared__ float xin[MEMD];
  __shared__ float hs[MEMD];
  const int m = threadIdx.x;
  xin[m] = inputs[m];
  hs[m]  = hsum[m];
  __syncthreads();

  float ai = bix[m] + bih[m];
  float au = bux[m] + buh[m];
  float ao = box[m] + boh[m];
  const float* wi = Wix + m * MEMD;  const float* vi = Wih + m * MEMD;
  const float* wu = Wux + m * MEMD;  const float* vu = Wuh + m * MEMD;
  const float* wo = Wox + m * MEMD;  const float* vo = Woh + m * MEMD;
#pragma unroll 4
  for (int k = 0; k < MEMD; ++k) {
    const float x = xin[k], h = hs[k];
    ai += x * wi[k] + h * vi[k];
    au += x * wu[k] + h * vu[k];
    ao += x * wo[k] + h * vo[k];
  }

  float S = 0.0f;
  for (int b = 0; b < nblk; ++b) S += partial[(long)b * MEMD + m];

  const float i = sigmoidf(ai);
  const float o = sigmoidf(ao);
  const float u = tanhf(au);
  const float c = i * u + S;
  const float h = o * tanhf(c);
  out[m] = c;
  out[MEMD + m] = h;
}

// ---------------------------------------------------------------------------
extern "C" void kernel_launch(void* const* d_in, const int* in_sizes, int n_in,
                              void* d_out, int out_size, void* d_ws, size_t ws_size,
                              hipStream_t stream) {
  const float* inputs  = (const float*)d_in[0];
  const float* child_c = (const float*)d_in[1];
  const float* child_h = (const float*)d_in[2];
  const float* hsum    = (const float*)d_in[3];
  const float* Wix = (const float*)d_in[4];   const float* bix = (const float*)d_in[5];
  const float* Wfx = (const float*)d_in[6];   const float* bfx = (const float*)d_in[7];
  const float* Wux = (const float*)d_in[8];   const float* bux = (const float*)d_in[9];
  const float* Wox = (const float*)d_in[10];  const float* box = (const float*)d_in[11];
  const float* Wih = (const float*)d_in[12];  const float* bih = (const float*)d_in[13];
  const float* Wfh = (const float*)d_in[14];  const float* bfh = (const float*)d_in[15];
  const float* Wuh = (const float*)d_in[16];  const float* buh = (const float*)d_in[17];
  const float* Woh = (const float*)d_in[18];  const float* boh = (const float*)d_in[19];

  const int nch  = in_sizes[1] / MEMD;              // 100000
  const int nblk = (nch + ROWS - 1) / ROWS;         // 391

  float* gfx     = (float*)d_ws;                    // [256]
  float* partial = gfx + MEMD;                      // [nblk][256]
  float* outp    = (float*)d_out;                   // c[256] ++ h[256]

  tree_lstm_gfx_kernel<<<1, 256, 0, stream>>>(inputs, Wfx, bfx, bfh, gfx);

  const size_t lds_bytes = (size_t)MEMD * KP * sizeof(unsigned short) // bf16 Wfh, padded
                         + (size_t)NWAVE * MEMD * sizeof(float)       // wave partials
                         + (size_t)MEMD * sizeof(float);              // gfx copy
  tree_lstm_forget_kernel<<<nblk, 256, lds_bytes, stream>>>(
      child_h, child_c, Wfh, gfx, partial, nch);

  tree_lstm_final_kernel<<<1, 256, 0, stream>>>(
      inputs, hsum, Wix, bix, Wux, bux, Wox, box,
      Wih, bih, Wuh, buh, Woh, boh, partial, nblk, outp);
}
